// GAE_82592221102104
// MI455X (gfx1250) — matrix operand order repaired and checked by reading, not compiled, not run on hardware
//
#include <hip/hip_runtime.h>
#include <hip/hip_bf16.h>
#include <stdint.h>

// Problem constants (match reference)
#define NN0   16384
#define FIN   384
#define HID   512
#define EE0   131072
#define NN1   4096      // 16384 * 0.25
#define NN2   1024      // 4096 * 0.25
#define KG1   6         // knn graph k at level 1 (START_K)
#define KIN   6         // INTERP_K

typedef unsigned short u16;
typedef u16    u16x8   __attribute__((ext_vector_type(8)));
typedef u16    u16x16  __attribute__((ext_vector_type(16)));
typedef __bf16 bf16x16 __attribute__((ext_vector_type(16)));
typedef float  f32x8   __attribute__((ext_vector_type(8)));

// ---------------------------------------------------------------------------
// Elementwise helpers
// ---------------------------------------------------------------------------
__global__ void k_fill(float* p, float v, long n) {
  long i = (long)blockIdx.x * blockDim.x + threadIdx.x;
  if (i < n) p[i] = v;
}

// f32 -> bf16 (round to nearest even), stored as raw u16 (row-major, for A)
__global__ void k_f32_to_bf16(const float* __restrict__ in, u16* __restrict__ out, long n) {
  long i = (long)blockIdx.x * blockDim.x + threadIdx.x;
  if (i < n) {
    unsigned u = __builtin_bit_cast(unsigned, in[i]);
    unsigned r = u + 0x7FFFu + ((u >> 16) & 1u);
    out[i] = (u16)(r >> 16);
  }
}

// Pack f32 weight W[K,N] row-major into bf16 WMMA B-fragment order:
// Wp[(((tn*Ks + ks)*32 + lane)*16 + j] = bf16(W[((j&7) + (lane>>4)*8 + (j>>3)*16 + ks*32)*N + tn*16 + (lane&15)])
// so each lane's 16-half B fragment is one contiguous 32-byte run.
__global__ void k_pack_w(const float* __restrict__ W, u16* __restrict__ Wp, int N, int K) {
  long t = (long)blockIdx.x * blockDim.x + threadIdx.x;
  if (t >= (long)N * K) return;
  const int Ks   = K >> 5;
  const int j    = (int)(t & 15);
  const int lane = (int)((t >> 4) & 31);
  const long rest = t >> 9;
  const int ks = (int)(rest % Ks);
  const int tn = (int)(rest / Ks);
  const int kk  = (j & 7) + (lane >> 4) * 8 + (j >> 3) * 16 + ks * 32;
  const int col = tn * 16 + (lane & 15);
  unsigned u = __builtin_bit_cast(unsigned, W[(size_t)kk * N + col]);
  unsigned r = u + 0x7FFFu + ((u >> 16) & 1u);
  Wp[t] = (u16)(r >> 16);
}

__global__ void k_bias_add(const float* __restrict__ in, const float* __restrict__ bias,
                           float* __restrict__ out, long n, int F) {
  long i = (long)blockIdx.x * blockDim.x + threadIdx.x;
  if (i < n) out[i] = in[i] + bias[(int)(i % F)];
}

__global__ void k_gather_rows(const float* __restrict__ in, const int* __restrict__ perm,
                              float* __restrict__ out, int n, int F) {
  long i = (long)blockIdx.x * blockDim.x + threadIdx.x;
  if (i < (long)n * F) {
    int r = (int)(i / F), f = (int)(i % F);
    out[i] = in[(size_t)perm[r] * F + f];
  }
}

// ---------------------------------------------------------------------------
// bf16 WMMA GEMM: C[M,N] = A[M,K] * B[K,N].
// A row-major bf16; Bp pre-packed fragment-order bf16 (k_pack_w); C f32.
// One 16(M) x 64(N) strip per wave: A fragment loaded as 2x b128 and reused
// across 4 v_wmma_f32_16x16x32_bf16; each B fragment is 2x b128 contiguous.
// Requires M%16==0, N%256==0, K%32==0 (true for all call sites).
// ---------------------------------------------------------------------------
__global__ void k_gemm_bf16(const u16* __restrict__ A, const u16* __restrict__ Bp,
                            float* __restrict__ C, int M, int N, int K) {
  const int lane  = threadIdx.x;                                  // 0..31
  const int grp   = lane >> 4;                                    // half-wave
  const int lid   = lane & 15;
  const int nt0   = (blockIdx.x * blockDim.y + threadIdx.y) * 4;  // first of 4 n-tiles
  const int tileM = blockIdx.y * 16;
  if (tileM >= M || nt0 * 16 >= N) return;  // wave-uniform -> EXEC all ones for WMMA
  const int Ks = K >> 5;

  f32x8 acc[4] = {};
  const u16* arow = A + (size_t)(tileM + lid) * K + grp * 8;
  for (int ks = 0; ks < Ks; ++ks) {
    // A fragment (ISA 16-bit 16x32 layout): two contiguous 8-half runs per lane
    u16x8 alo = *(const u16x8*)(arow + ks * 32);
    u16x8 ahi = *(const u16x8*)(arow + ks * 32 + 16);
    __builtin_prefetch(arow + ks * 32 + 32, 0, 3);
    u16x16 au;
#pragma unroll
    for (int j = 0; j < 8; ++j) { au[j] = alo[j]; au[8 + j] = ahi[j]; }
    bf16x16 a = __builtin_bit_cast(bf16x16, au);
#pragma unroll
    for (int t = 0; t < 4; ++t) {
      const u16x16 bu =
          *(const u16x16*)(Bp + (((size_t)(nt0 + t) * Ks + ks) * 32 + lane) * 16);
      bf16x16 b = __builtin_bit_cast(bf16x16, bu);
      acc[t] = __builtin_amdgcn_wmma_f32_16x16x32_bf16(false, a, false, b, (short)0,
                                                       acc[t], false, false);
    }
  }
  // C/D layout: VGPR r -> row r (+8 for upper half-wave), lane%16 -> col
#pragma unroll
  for (int t = 0; t < 4; ++t) {
#pragma unroll
    for (int r = 0; r < 8; ++r) {
      C[(size_t)(tileM + r + grp * 8) * N + (nt0 + t) * 16 + lid] = acc[t][r];
    }
  }
}

// ---------------------------------------------------------------------------
// GCN pieces: deg = 2 + indegree; dinv = rsqrt(deg); coef[e]=dinv[src]*dinv[dst]
// agg[dst] += coef * h[src]  (float atomics); finish adds self term + bias (+relu)
// ---------------------------------------------------------------------------
__global__ void k_deg_count(const int* __restrict__ dst, long E, float* deg) {
  long e = (long)blockIdx.x * blockDim.x + threadIdx.x;
  if (e < E) atomicAdd(&deg[dst[e]], 1.0f);
}

__global__ void k_rsqrt(float* p, int n) {
  int i = blockIdx.x * blockDim.x + threadIdx.x;
  if (i < n) p[i] = rsqrtf(p[i]);
}

__global__ void k_edge_coef(const int* __restrict__ src, const int* __restrict__ dst,
                            const float* __restrict__ dinv, float* __restrict__ coef, long E) {
  long e = (long)blockIdx.x * blockDim.x + threadIdx.x;
  if (e < E) coef[e] = dinv[src[e]] * dinv[dst[e]];
}

__global__ void k_gcn_agg(const float* __restrict__ h, const int* __restrict__ src,
                          const int* __restrict__ dst, const float* __restrict__ coef,
                          float* __restrict__ agg, long E, int F) {
  long t = (long)blockIdx.x * blockDim.x + threadIdx.x;
  if (t < E * F) {
    long e = t / F;
    int  f = (int)(t % F);
    atomicAdd(&agg[(size_t)dst[e] * F + f], coef[e] * h[(size_t)src[e] * F + f]);
  }
}

__global__ void k_gcn_finish(float* __restrict__ agg, const float* __restrict__ h,
                             const float* __restrict__ dinv, const float* __restrict__ bias,
                             long n, int F, int relu) {
  long t = (long)blockIdx.x * blockDim.x + threadIdx.x;
  if (t < n * F) {
    int r = (int)(t / F), f = (int)(t % F);
    float d = dinv[r];
    float v = agg[t] + h[t] * (2.0f * d * d) + bias[f];
    agg[t] = relu ? fmaxf(v, 0.0f) : v;
  }
}

// ---------------------------------------------------------------------------
// Farthest point sampling: single 1024-thread workgroup, LDS tree argmax
// (first-max-index tie-break like jnp.argmax), deterministic start at node 0.
// ---------------------------------------------------------------------------
__global__ void k_fps(const float* __restrict__ pos, int n, int n_keep,
                      int* __restrict__ sel, float* __restrict__ dist) {
  __shared__ float sval[1024];
  __shared__ int   sidx[1024];
  __shared__ float bx, by;
  const int tid = threadIdx.x;
  const float p0x = pos[0], p0y = pos[1];
  for (int j = tid; j < n; j += blockDim.x) {
    float dx = pos[2 * j] - p0x, dy = pos[2 * j + 1] - p0y;
    dist[j] = dx * dx + dy * dy;
  }
  if (tid == 0) sel[0] = 0;
  __syncthreads();
  for (int i = 1; i < n_keep; ++i) {
    float best = -1.0f; int bi = 0x7FFFFFFF;
    for (int j = tid; j < n; j += blockDim.x) {
      float d = dist[j];
      if (d > best) { best = d; bi = j; }
    }
    sval[tid] = best; sidx[tid] = bi;
    __syncthreads();
    for (int s = blockDim.x >> 1; s > 0; s >>= 1) {
      if (tid < s) {
        if (sval[tid + s] > sval[tid] ||
            (sval[tid + s] == sval[tid] && sidx[tid + s] < sidx[tid])) {
          sval[tid] = sval[tid + s]; sidx[tid] = sidx[tid + s];
        }
      }
      __syncthreads();
    }
    if (tid == 0) {
      int idx = sidx[0];
      sel[i] = idx;
      bx = pos[2 * idx]; by = pos[2 * idx + 1];
    }
    __syncthreads();
    const float qx = bx, qy = by;
    for (int j = tid; j < n; j += blockDim.x) {
      float dx = pos[2 * j] - qx, dy = pos[2 * j + 1] - qy;
      float d = dx * dx + dy * dy;
      if (d < dist[j]) dist[j] = d;
    }
    __syncthreads();
  }
}

// ---------------------------------------------------------------------------
// kNN graph (self excluded): one thread per point, register insertion top-k
// src[i*k+kk] = kk-th nearest neighbor, dst[i*k+kk] = i (source_to_target flow)
// ---------------------------------------------------------------------------
__global__ void k_knn_graph(const float* __restrict__ pos, int n, int k,
                            int* __restrict__ src, int* __restrict__ dst) {
  int i = blockIdx.x * blockDim.x + threadIdx.x;
  if (i >= n) return;
  float bd[8]; int bi[8];
#pragma unroll
  for (int t = 0; t < 8; ++t) { bd[t] = 1e30f; bi[t] = -1; }
  const float px = pos[2 * i], py = pos[2 * i + 1];
  for (int j = 0; j < n; ++j) {
    if (j == i) continue;
    float dx = pos[2 * j] - px, dy = pos[2 * j + 1] - py;
    float d = dx * dx + dy * dy;
    if (d < bd[k - 1]) {
      int t = k - 1;
      while (t > 0 && bd[t - 1] > d) { bd[t] = bd[t - 1]; bi[t] = bi[t - 1]; --t; }
      bd[t] = d; bi[t] = j;
    }
  }
  for (int kk = 0; kk < k; ++kk) { src[i * k + kk] = bi[kk]; dst[i * k + kk] = i; }
}

// kNN query for interpolation (no self-exclusion): idx + weight 1/max(d2,1e-16)
__global__ void k_knn_query(const float* __restrict__ posy, int ny,
                            const float* __restrict__ posx, int nx, int k,
                            int* __restrict__ idx, float* __restrict__ w) {
  int i = blockIdx.x * blockDim.x + threadIdx.x;
  if (i >= ny) return;
  float bd[8]; int bi[8];
#pragma unroll
  for (int t = 0; t < 8; ++t) { bd[t] = 1e30f; bi[t] = 0; }
  const float px = posy[2 * i], py = posy[2 * i + 1];
  for (int j = 0; j < nx; ++j) {
    float dx = posx[2 * j] - px, dy = posx[2 * j + 1] - py;
    float d = dx * dx + dy * dy;
    if (d < bd[k - 1]) {
      int t = k - 1;
      while (t > 0 && bd[t - 1] > d) { bd[t] = bd[t - 1]; bi[t] = bi[t - 1]; --t; }
      bd[t] = d; bi[t] = j;
    }
  }
  for (int kk = 0; kk < k; ++kk) {
    idx[i * k + kk] = bi[kk];
    w[i * k + kk]   = 1.0f / fmaxf(bd[kk], 1e-16f);
  }
}

__global__ void k_knn_apply(const float* __restrict__ x, const int* __restrict__ idx,
                            const float* __restrict__ w, float* __restrict__ out,
                            long ny, int F, int k) {
  long t = (long)blockIdx.x * blockDim.x + threadIdx.x;
  if (t < ny * F) {
    int y = (int)(t / F), f = (int)(t % F);
    float num = 0.0f, den = 0.0f;
    for (int kk = 0; kk < k; ++kk) {
      float ww = w[y * k + kk];
      num += ww * x[(size_t)idx[y * k + kk] * F + f];
      den += ww;
    }
    out[t] = num / den;
  }
}

// ---------------------------------------------------------------------------
// Orchestration
// ---------------------------------------------------------------------------
static inline unsigned cdiv(long a, long b) { return (unsigned)((a + b - 1) / b); }

extern "C" void kernel_launch(void* const* d_in, const int* in_sizes, int n_in,
                              void* d_out, int out_size, void* d_ws, size_t ws_size,
                              hipStream_t stream) {
  (void)in_sizes; (void)n_in; (void)out_size; (void)ws_size;
  const float* x   = (const float*)d_in[0];
  const float* pos = (const float*)d_in[1];
  const int*   ei  = (const int*)d_in[2];          // (2, E0): src then dst
  const float* Wd0 = (const float*)d_in[3];  const float* bd0 = (const float*)d_in[4];
  const float* Wd1 = (const float*)d_in[5];  const float* bd1 = (const float*)d_in[6];
  const float* Wu0 = (const float*)d_in[7];  const float* bu0 = (const float*)d_in[8];
  const float* Wu1 = (const float*)d_in[9];  const float* bu1 = (const float*)d_in[10];
  const float* Wl  = (const float*)d_in[11]; const float* bl  = (const float*)d_in[12];
  const int* src0 = ei;
  const int* dst0 = ei + EE0;

  // ---- workspace bump allocator ----
  char* ws = (char*)d_ws;
  size_t off = 0;
  auto alloc = [&](size_t bytes) -> void* {
    off = (off + 255) & ~(size_t)255;
    void* p = ws + off;
    off += bytes;
    return p;
  };
  u16* wd0b = (u16*)alloc((size_t)FIN * HID * 2);
  u16* wd1b = (u16*)alloc((size_t)HID * HID * 2);
  u16* wu0b = (u16*)alloc((size_t)HID * HID * 2);
  u16* wu1b = (u16*)alloc((size_t)HID * HID * 2);
  u16* wlb  = (u16*)alloc((size_t)HID * HID * 2);
  u16*   xb    = (u16*)alloc((size_t)NN0 * HID * 2);    // bf16 activation staging (A)
  float* hbuf  = (float*)alloc((size_t)NN0 * HID * 4);  // GEMM outputs
  float* agg   = (float*)alloc((size_t)NN0 * HID * 4);  // aggregation / GCN result
  float* act   = (float*)alloc((size_t)NN0 * HID * 4);  // gather / interp staging
  float* dinv0 = (float*)alloc((size_t)NN0 * 4);
  float* dinv1 = (float*)alloc((size_t)NN1 * 4);
  float* coef  = (float*)alloc((size_t)EE0 * 4);
  float* dist  = (float*)alloc((size_t)NN0 * 4);
  int*   perm1 = (int*)alloc((size_t)NN1 * 4);
  int*   perm2 = (int*)alloc((size_t)NN2 * 4);
  float* pos1  = (float*)alloc((size_t)NN1 * 2 * 4);
  float* pos2  = (float*)alloc((size_t)NN2 * 2 * 4);
  int*   src1  = (int*)alloc((size_t)NN1 * KG1 * 4);
  int*   dst1  = (int*)alloc((size_t)NN1 * KG1 * 4);
  int*   iidx  = (int*)alloc((size_t)NN0 * KIN * 4);
  float* iw    = (float*)alloc((size_t)NN0 * KIN * 4);

  const int TPB = 256;
  const dim3 gblock(32, 4);  // 4 waves per block, one 16x64 strip each
#define L1D(n) dim3(cdiv((long)(n), TPB)), dim3(TPB), 0, stream
#define GEMM_GRID(M) dim3(HID / 256, (M) / 16)

  // ---- pack weights to bf16 fragment order once ----
  k_pack_w<<<L1D((long)FIN * HID)>>>(Wd0, wd0b, HID, FIN);
  k_pack_w<<<L1D((long)HID * HID)>>>(Wd1, wd1b, HID, HID);
  k_pack_w<<<L1D((long)HID * HID)>>>(Wu0, wu0b, HID, HID);
  k_pack_w<<<L1D((long)HID * HID)>>>(Wu1, wu1b, HID, HID);
  k_pack_w<<<L1D((long)HID * HID)>>>(Wl,  wlb,  HID, HID);

  // ================= down level 0: GCN(x, orig edges) + relu =================
  k_f32_to_bf16<<<L1D((long)NN0 * FIN)>>>(x, xb, (long)NN0 * FIN);
  k_gemm_bf16<<<GEMM_GRID(NN0), gblock, 0, stream>>>(xb, wd0b, hbuf, NN0, HID, FIN);
  k_fill<<<L1D(NN0)>>>(dinv0, 2.0f, NN0);
  k_deg_count<<<L1D(EE0)>>>(dst0, EE0, dinv0);
  k_rsqrt<<<L1D(NN0)>>>(dinv0, NN0);
  k_edge_coef<<<L1D(EE0)>>>(src0, dst0, dinv0, coef, EE0);
  k_fill<<<L1D((long)NN0 * HID)>>>(agg, 0.0f, (long)NN0 * HID);
  k_gcn_agg<<<L1D((long)EE0 * HID)>>>(hbuf, src0, dst0, coef, agg, EE0, HID);
  k_gcn_finish<<<L1D((long)NN0 * HID)>>>(agg, hbuf, dinv0, bd0, NN0, HID, 1);

  // ---- pool 1: FPS -> 4096, gather pos/x, build knn graph (k=6) ----
  k_fps<<<1, 1024, 0, stream>>>(pos, NN0, NN1, perm1, dist);
  k_gather_rows<<<L1D((long)NN1 * 2)>>>(pos, perm1, pos1, NN1, 2);
  k_gather_rows<<<L1D((long)NN1 * HID)>>>(agg, perm1, act, NN1, HID);
  k_knn_graph<<<L1D(NN1)>>>(pos1, NN1, KG1, src1, dst1);

  // ================= down level 1: GCN(x1, knn1 edges) + relu ================
  k_f32_to_bf16<<<L1D((long)NN1 * HID)>>>(act, xb, (long)NN1 * HID);
  k_gemm_bf16<<<GEMM_GRID(NN1), gblock, 0, stream>>>(xb, wd1b, hbuf, NN1, HID, HID);
  k_fill<<<L1D(NN1)>>>(dinv1, 2.0f, NN1);
  k_deg_count<<<L1D((long)NN1 * KG1)>>>(dst1, (long)NN1 * KG1, dinv1);
  k_rsqrt<<<L1D(NN1)>>>(dinv1, NN1);
  k_edge_coef<<<L1D((long)NN1 * KG1)>>>(src1, dst1, dinv1, coef, (long)NN1 * KG1);
  k_fill<<<L1D((long)NN1 * HID)>>>(agg, 0.0f, (long)NN1 * HID);
  k_gcn_agg<<<L1D((long)NN1 * KG1 * HID)>>>(hbuf, src1, dst1, coef, agg, (long)NN1 * KG1, HID);
  k_gcn_finish<<<L1D((long)NN1 * HID)>>>(agg, hbuf, dinv1, bd1, NN1, HID, 1);

  // ---- pool 2: FPS -> 1024, gather pos/x (level-2 knn graph is dead code) ----
  k_fps<<<1, 1024, 0, stream>>>(pos1, NN1, NN2, perm2, dist);
  k_gather_rows<<<L1D((long)NN2 * 2)>>>(pos1, perm2, pos2, NN2, 2);
  k_gather_rows<<<L1D((long)NN2 * HID)>>>(agg, perm2, act, NN2, HID);

  // ================= up 0: interp(pos2 -> pos1), GCN(knn1 edges) + relu ======
  k_knn_query<<<L1D(NN1)>>>(pos1, NN1, pos2, NN2, KIN, iidx, iw);
  k_knn_apply<<<L1D((long)NN1 * HID)>>>(act, iidx, iw, agg, (long)NN1, HID, KIN);
  k_f32_to_bf16<<<L1D((long)NN1 * HID)>>>(agg, xb, (long)NN1 * HID);
  k_gemm_bf16<<<GEMM_GRID(NN1), gblock, 0, stream>>>(xb, wu0b, hbuf, NN1, HID, HID);
  k_fill<<<L1D((long)NN1 * HID)>>>(agg, 0.0f, (long)NN1 * HID);
  k_gcn_agg<<<L1D((long)NN1 * KG1 * HID)>>>(hbuf, src1, dst1, coef, agg, (long)NN1 * KG1, HID);
  k_gcn_finish<<<L1D((long)NN1 * HID)>>>(agg, hbuf, dinv1, bu0, NN1, HID, 1);

  // ================= up 1: interp(pos1 -> pos0), GCN(orig edges) + relu ======
  k_knn_query<<<L1D(NN0)>>>(pos, NN0, pos1, NN1, KIN, iidx, iw);
  k_knn_apply<<<L1D((long)NN0 * HID)>>>(agg, iidx, iw, act, (long)NN0, HID, KIN);
  k_f32_to_bf16<<<L1D((long)NN0 * HID)>>>(act, xb, (long)NN0 * HID);
  k_gemm_bf16<<<GEMM_GRID(NN0), gblock, 0, stream>>>(xb, wu1b, hbuf, NN0, HID, HID);
  k_edge_coef<<<L1D(EE0)>>>(src0, dst0, dinv0, coef, EE0);  // coef was clobbered at level 1
  k_fill<<<L1D((long)NN0 * HID)>>>(agg, 0.0f, (long)NN0 * HID);
  k_gcn_agg<<<L1D((long)EE0 * HID)>>>(hbuf, src0, dst0, coef, agg, EE0, HID);
  k_gcn_finish<<<L1D((long)NN0 * HID)>>>(agg, hbuf, dinv0, bu1, NN0, HID, 1);

  // ================= final linear =================
  k_f32_to_bf16<<<L1D((long)NN0 * HID)>>>(agg, xb, (long)NN0 * HID);
  k_gemm_bf16<<<GEMM_GRID(NN0), gblock, 0, stream>>>(xb, wlb, hbuf, NN0, HID, HID);
  k_bias_add<<<L1D((long)NN0 * HID)>>>(hbuf, bl, (float*)d_out, (long)NN0 * HID, HID);
#undef GEMM_GRID
#undef L1D
}
